// ActorCritic_7327214207032
// MI455X (gfx1250) — matrix operand order repaired and checked by reading, not compiled, not run on hardware
//
#include <hip/hip_runtime.h>
#include <stdint.h>

#define N_NODES 50000
#define N_EDGES 800000
#define IN_FEATS 128
#define HID 64
#define HEADS 4
#define ACTIONS 16
#define HD 256   // HEADS*HID

typedef __attribute__((ext_vector_type(16))) __bf16 v16bf;
typedef __attribute__((ext_vector_type(8)))  float  v8f;

union AFrag {
    v16bf v;
    unsigned short u[16];
    uint4 q[2];
};

__device__ __forceinline__ unsigned short f2bf(float f) {
    __bf16 b = (__bf16)f;                       // hardware cvt on gfx1250
    return __builtin_bit_cast(unsigned short, b);
}
__device__ __forceinline__ unsigned pk2bf(float lo, float hi) {
    return (unsigned)f2bf(lo) | ((unsigned)f2bf(hi) << 16);
}
__device__ __forceinline__ unsigned flipf(float f) {
    unsigned u = __float_as_uint(f);
    return (u & 0x80000000u) ? ~u : (u | 0x80000000u);
}
__device__ __forceinline__ float unflipf(unsigned k) {
    unsigned u = (k & 0x80000000u) ? (k & 0x7fffffffu) : ~k;
    return __uint_as_float(u);
}

// ---------------- K0a: weight preconversion into per-lane WMMA B-fragment layout.
// fcwb:  [colTile(16)][kk(4)][lane(32)][16]   from fc_w  [128x256]
// wcatb: [colTile(8)][kk(8)][lane(32)][16]    from Wcat  [256x128] (a_w1 split)
// aw2b:  [kk(2)][lane(32)][16]                from a_w2  [64x16]
__global__ void k_prep(const float* __restrict__ fcw, const float* __restrict__ aw1,
                       const float* __restrict__ aw2, unsigned short* __restrict__ fcwb,
                       unsigned short* __restrict__ wcatb, unsigned short* __restrict__ aw2b) {
    int tid = blockIdx.x * 256 + threadIdx.x;
    if (tid < 32768) {                                   // fcwb
        int j = tid & 15, lane = (tid >> 4) & 31, kk = (tid >> 9) & 3, ct = tid >> 11;
        int sel = lane >> 4, l16 = lane & 15;
        int k = kk * 32 + sel * 8 + (j & 7) + (j >> 3) * 16;
        int n = ct * 16 + l16;
        fcwb[tid] = f2bf(fcw[k * HD + n]);
    } else if (tid < 65536) {                            // wcatb
        int t2 = tid - 32768;
        int j = t2 & 15, lane = (t2 >> 4) & 31, kk = (t2 >> 9) & 7, ct = t2 >> 12;
        int sel = lane >> 4, l16 = lane & 15;
        int k = kk * 32 + sel * 8 + (j & 7) + (j >> 3) * 16;
        int jj = ct * 16 + l16;                          // 0..127
        float v = (jj < 64) ? aw1[k * 64 + jj] : aw1[(256 + k) * 64 + (jj - 64)];
        wcatb[t2] = f2bf(v);
    } else if (tid < 65536 + 1024) {                     // aw2b
        int t2 = tid - 65536;
        int j = t2 & 15, lane = (t2 >> 4) & 31, kk = t2 >> 9;
        int sel = lane >> 4, l16 = lane & 15;
        int k = kk * 32 + sel * 8 + (j & 7) + (j >> 3) * 16;
        aw2b[t2] = f2bf(aw2[k * ACTIONS + l16]);
    }
}

// ---------------- K0b: x fp32 -> packed bf16 (one-time; x is reused by 16 col tiles)
__global__ void k_cvt_x(const float* __restrict__ x, unsigned short* __restrict__ xb) {
    int tid = blockIdx.x * 256 + threadIdx.x;           // 1,600,000 exact
    float4 v = *(const float4*)(x + tid * 4);
    uint2 pk;
    pk.x = pk2bf(v.x, v.y);
    pk.y = pk2bf(v.z, v.w);
    *(uint2*)(xb + tid * 4) = pk;
}

// ---------------- K1: h = x @ fc_w  [50000x128]x[128x256], bf16 WMMA, fp32 acc
__global__ __launch_bounds__(128) void k_gemm_fc(const unsigned short* __restrict__ xb,
                                                 const unsigned short* __restrict__ fcwb,
                                                 float* __restrict__ h) {
    int wave = threadIdx.x >> 5, lane = threadIdx.x & 31;
    int tile = blockIdx.x * 4 + wave;       // 50000 tiles total
    int rowTile = tile >> 4;                // 0..3124
    int colTile = tile & 15;                // 0..15
    int sel = lane >> 4, l16 = lane & 15;
    int M = rowTile * 16 + l16;
    int colN = colTile * 16 + l16;
    v8f c = {};
#pragma unroll
    for (int kk = 0; kk < 4; ++kk) {
        AFrag a, b;
        const unsigned short* ar = xb + M * IN_FEATS + kk * 32 + sel * 8;
        a.q[0] = *(const uint4*)(ar);
        a.q[1] = *(const uint4*)(ar + 16);
        const unsigned short* bp = fcwb + ((colTile * 4 + kk) * 32 + lane) * 16;
        b.q[0] = *(const uint4*)(bp);
        b.q[1] = *(const uint4*)(bp + 8);
        c = __builtin_amdgcn_wmma_f32_16x16x32_bf16(false, a.v, false, b.v, (short)0, c, false, false);
    }
#pragma unroll
    for (int r = 0; r < 8; ++r)
        h[(rowTile * 16 + sel * 8 + r) * HD + colN] = c[r];
}

// ---------------- K2: el/er per (node, head)
__global__ void k_attn(const float* __restrict__ h, const float* __restrict__ attl,
                       const float* __restrict__ attr, float* __restrict__ el,
                       float* __restrict__ er) {
    int tid = blockIdx.x * 256 + threadIdx.x;
    if (tid >= N_NODES * HEADS) return;
    int n = tid >> 2, hh = tid & 3;
    const float* hp = h + n * HD + hh * HID;
    const float* lp = attl + hh * HID;
    const float* rp = attr + hh * HID;
    float sl = 0.f, sr = 0.f;
#pragma unroll
    for (int i = 0; i < 16; ++i) {
        float4 hv = *(const float4*)(hp + i * 4);
        float4 lv = *(const float4*)(lp + i * 4);
        float4 rv = *(const float4*)(rp + i * 4);
        sl += hv.x * lv.x + hv.y * lv.y + hv.z * lv.z + hv.w * lv.w;
        sr += hv.x * rv.x + hv.y * rv.y + hv.z * rv.z + hv.w * rv.w;
    }
    el[tid] = sl;
    er[tid] = sr;
}

// ---------------- K3: e = leaky_relu(el[src]+er[dst]); atomic segment max
__global__ void k_edge_e(const int* __restrict__ src, const int* __restrict__ dst,
                         const float* __restrict__ el, const float* __restrict__ er,
                         float* __restrict__ ews, unsigned* __restrict__ m) {
    int e = blockIdx.x * 256 + threadIdx.x;   // 800000 exact
    int s = src[e], t = dst[e];
#pragma unroll
    for (int hh = 0; hh < 4; ++hh) {
        float v = el[s * 4 + hh] + er[t * 4 + hh];
        v = (v > 0.f) ? v : 0.2f * v;
        ews[e * 4 + hh] = v;
        atomicMax(m + t * 4 + hh, flipf(v));
    }
}

// ---------------- K4: ex = exp(e - m[dst]); z[dst] += ex
__global__ void k_edge_exp(const int* __restrict__ dst, float* __restrict__ ews,
                           const unsigned* __restrict__ m, float* __restrict__ z) {
    int e = blockIdx.x * 256 + threadIdx.x;
    int t = dst[e];
#pragma unroll
    for (int hh = 0; hh < 4; ++hh) {
        float mx = unflipf(m[t * 4 + hh]);
        float xv = __expf(ews[e * 4 + hh] - mx);
        ews[e * 4 + hh] = xv;
        atomicAdd(z + t * 4 + hh, xv);
    }
}

// ---------------- K5: rst[dst] += (ex/z[dst]) * h[src]   (wave per edge)
__global__ __launch_bounds__(256) void k_msg(const int* __restrict__ src,
                                             const int* __restrict__ dst,
                                             const float* __restrict__ ews,
                                             const float* __restrict__ z,
                                             const float* __restrict__ h,
                                             float* __restrict__ rst) {
    int wave = threadIdx.x >> 5, lane = threadIdx.x & 31;
    int e = blockIdx.x * 8 + wave;            // 800000 exact
    int s = src[e], t = dst[e];
#pragma unroll
    for (int i = 0; i < 8; ++i) {
        int cidx = lane + i * 32;
        int hh = cidx >> 6;
        float alpha = ews[e * 4 + hh] / z[t * 4 + hh];
        atomicAdd(&rst[t * HD + cidx], alpha * h[s * HD + cidx]);
    }
}

// ---------------- K6: h2 = elu(rst + bias) in place + packed bf16 copy
__global__ void k_elu(float* __restrict__ rst, const float* __restrict__ gbias,
                      unsigned short* __restrict__ h2b) {
    int tid = blockIdx.x * 256 + threadIdx.x;  // 3.2M exact, 4 elems each
    int idx = tid * 4;
    int cidx = idx & 255;
    float4 v = *(const float4*)(rst + idx);
    float4 g = *(const float4*)(gbias + cidx);
    v.x += g.x; v.y += g.y; v.z += g.z; v.w += g.w;
    v.x = v.x > 0.f ? v.x : (__expf(v.x) - 1.f);
    v.y = v.y > 0.f ? v.y : (__expf(v.y) - 1.f);
    v.z = v.z > 0.f ? v.z : (__expf(v.z) - 1.f);
    v.w = v.w > 0.f ? v.w : (__expf(v.w) - 1.f);
    *(float4*)(rst + idx) = v;
    uint2 pk;
    pk.x = pk2bf(v.x, v.y);
    pk.y = pk2bf(v.z, v.w);
    *(uint2*)(h2b + idx) = pk;
}

// ---------------- K6b: graph_rep = mean over nodes, per column
__global__ void k_colmean(const float* __restrict__ h2, float* __restrict__ gr) {
    __shared__ float sbuf[256];
    int cidx = blockIdx.x;
    float acc = 0.f;
    for (int n = threadIdx.x; n < N_NODES; n += 256) acc += h2[n * HD + cidx];
    sbuf[threadIdx.x] = acc;
    __syncthreads();
    for (int s2 = 128; s2 > 0; s2 >>= 1) {
        if (threadIdx.x < s2) sbuf[threadIdx.x] += sbuf[threadIdx.x + s2];
        __syncthreads();
    }
    if (threadIdx.x == 0) gr[cidx] = sbuf[0] * (1.0f / N_NODES);
}

// ---------------- K7: hsd = h2 @ wcat  [50000x256]x[256x128], bf16 WMMA
__global__ __launch_bounds__(128) void k_gemm_hsd(const unsigned short* __restrict__ h2b,
                                                  const unsigned short* __restrict__ wcatb,
                                                  float* __restrict__ hsd) {
    int wave = threadIdx.x >> 5, lane = threadIdx.x & 31;
    int tile = blockIdx.x * 4 + wave;       // 25000 tiles
    int rowTile = tile >> 3;
    int colTile = tile & 7;
    int sel = lane >> 4, l16 = lane & 15;
    int M = rowTile * 16 + l16;
    int colN = colTile * 16 + l16;
    v8f c = {};
#pragma unroll
    for (int kk = 0; kk < 8; ++kk) {
        AFrag a, b;
        const unsigned short* ar = h2b + M * HD + kk * 32 + sel * 8;
        a.q[0] = *(const uint4*)(ar);
        a.q[1] = *(const uint4*)(ar + 16);
        const unsigned short* bp = wcatb + ((colTile * 8 + kk) * 32 + lane) * 16;
        b.q[0] = *(const uint4*)(bp);
        b.q[1] = *(const uint4*)(bp + 8);
        c = __builtin_amdgcn_wmma_f32_16x16x32_bf16(false, a.v, false, b.v, (short)0, c, false, false);
    }
#pragma unroll
    for (int r = 0; r < 8; ++r)
        hsd[(rowTile * 16 + sel * 8 + r) * 128 + colN] = c[r];
}

// ---------------- K8: per-edge actor. R=relu(hs[src]+hd[dst]+b1) (16 edges/wave),
//                  then R[16x64] @ a_w2[64x16] via WMMA, +b2 -> logits
__global__ __launch_bounds__(256) void k_edge_actor(const int* __restrict__ src,
                                                    const int* __restrict__ dst,
                                                    const float* __restrict__ hsd,
                                                    const unsigned short* __restrict__ aw2b,
                                                    const float* __restrict__ ab1,
                                                    const float* __restrict__ ab2,
                                                    float* __restrict__ out) {
    __shared__ unsigned short R[8][16][64];
    int wave = threadIdx.x >> 5, lane = threadIdx.x & 31;
    int tile = blockIdx.x * 8 + wave;       // 50000 tiles exact
    int edgeBase = tile * 16;
    // phase 1: 2 lanes per edge, each covers 32 of 64 hidden dims
    int m = lane >> 1, hf = lane & 1;
    int e = edgeBase + m;
    int s = src[e], t = dst[e];
    const float* ps = hsd + s * 128 + hf * 32;
    const float* pt = hsd + t * 128 + 64 + hf * 32;
    const float* pb = ab1 + hf * 32;
#pragma unroll
    for (int i = 0; i < 8; ++i) {
        float4 a4 = *(const float4*)(ps + i * 4);
        float4 b4 = *(const float4*)(pt + i * 4);
        float4 c4 = *(const float4*)(pb + i * 4);
        float r0 = a4.x + b4.x + c4.x; r0 = r0 > 0.f ? r0 : 0.f;
        float r1 = a4.y + b4.y + c4.y; r1 = r1 > 0.f ? r1 : 0.f;
        float r2 = a4.z + b4.z + c4.z; r2 = r2 > 0.f ? r2 : 0.f;
        float r3 = a4.w + b4.w + c4.w; r3 = r3 > 0.f ? r3 : 0.f;
        uint2 pk;
        pk.x = pk2bf(r0, r1);
        pk.y = pk2bf(r2, r3);
        *(uint2*)&R[wave][m][hf * 32 + i * 4] = pk;
    }
    __syncthreads();
    // phase 2: WMMA  R[16x64] x aw2[64x16]
    int sel = lane >> 4, l16 = lane & 15;
    v8f c = {};
#pragma unroll
    for (int kk = 0; kk < 2; ++kk) {
        AFrag a, b;
        const unsigned short* row = &R[wave][l16][0];
        int kb = kk * 32 + sel * 8;
        a.q[0] = *(const uint4*)(row + kb);
        a.q[1] = *(const uint4*)(row + kb + 16);
        const unsigned short* bp = aw2b + (kk * 32 + lane) * 16;
        b.q[0] = *(const uint4*)(bp);
        b.q[1] = *(const uint4*)(bp + 8);
        c = __builtin_amdgcn_wmma_f32_16x16x32_bf16(false, a.v, false, b.v, (short)0, c, false, false);
    }
    float bias = ab2[l16];
#pragma unroll
    for (int r = 0; r < 8; ++r)
        out[(edgeBase + sel * 8 + r) * ACTIONS + l16] = c[r] + bias;
}

// ---------------- K9: critic MLP, single block
__global__ void k_critic(const float* __restrict__ gr, const float* __restrict__ cw1,
                         const float* __restrict__ cb1, const float* __restrict__ cw2,
                         const float* __restrict__ cb2, float* __restrict__ out) {
    __shared__ float hid[64];
    int t = threadIdx.x;
    float acc = cb1[t];
    for (int k = 0; k < HD; ++k) acc += gr[k] * cw1[k * HID + t];
    hid[t] = acc > 0.f ? acc : 0.f;
    __syncthreads();
    if (t == 0) {
        float sv = cb2[0];
        for (int u = 0; u < HID; ++u) sv += hid[u] * cw2[u];
        out[(size_t)N_EDGES * ACTIONS] = sv;
    }
}

// ---------------- workspace layout (bytes, 256-aligned)
static constexpr size_t H_OFF     = 0;                       // h        50000*256*4
static constexpr size_t RST_OFF   = 51200000;                // rst/h2   50000*256*4
static constexpr size_t EL_OFF    = 102400000;               // el       50000*4*4
static constexpr size_t ER_OFF    = 103200000;               // er
static constexpr size_t M_OFF     = 104000000;               // m keys   50000*4*4
static constexpr size_t Z_OFF     = 104800000;               // z
static constexpr size_t E_OFF     = 105600000;               // e/ex     800000*4*4
static constexpr size_t HSD_OFF   = 118400000;               // hsd      50000*128*4
static constexpr size_t XB_OFF    = 144000000;               // x bf16   50000*128*2
static constexpr size_t H2B_OFF   = 156800000;               // h2 bf16  50000*256*2
static constexpr size_t FCWB_OFF  = 182400000;               // fcwb     32768*2
static constexpr size_t WCATB_OFF = 182465536;               // wcatb    32768*2
static constexpr size_t AW2B_OFF  = 182531072;               // aw2b     1024*2
static constexpr size_t GR_OFF    = 182533120;               // graph_rep 256*4

extern "C" void kernel_launch(void* const* d_in, const int* in_sizes, int n_in,
                              void* d_out, int out_size, void* d_ws, size_t ws_size,
                              hipStream_t stream) {
    (void)in_sizes; (void)n_in; (void)out_size; (void)ws_size;
    const float* x    = (const float*)d_in[0];
    const int*   src  = (const int*)d_in[1];
    const int*   dst  = (const int*)d_in[2];
    const float* fcw  = (const float*)d_in[3];
    const float* attl = (const float*)d_in[4];
    const float* attr = (const float*)d_in[5];
    const float* gbias= (const float*)d_in[6];
    const float* aw1  = (const float*)d_in[7];
    const float* ab1  = (const float*)d_in[8];
    const float* aw2  = (const float*)d_in[9];
    const float* ab2  = (const float*)d_in[10];
    const float* cw1  = (const float*)d_in[11];
    const float* cb1  = (const float*)d_in[12];
    const float* cw2  = (const float*)d_in[13];
    const float* cb2  = (const float*)d_in[14];

    char* ws = (char*)d_ws;
    float*          h     = (float*)(ws + H_OFF);
    float*          rst   = (float*)(ws + RST_OFF);
    float*          el    = (float*)(ws + EL_OFF);
    float*          er    = (float*)(ws + ER_OFF);
    unsigned*       mk    = (unsigned*)(ws + M_OFF);
    float*          z     = (float*)(ws + Z_OFF);
    float*          ews   = (float*)(ws + E_OFF);
    float*          hsd   = (float*)(ws + HSD_OFF);
    unsigned short* xb    = (unsigned short*)(ws + XB_OFF);
    unsigned short* h2b   = (unsigned short*)(ws + H2B_OFF);
    unsigned short* fcwb  = (unsigned short*)(ws + FCWB_OFF);
    unsigned short* wcatb = (unsigned short*)(ws + WCATB_OFF);
    unsigned short* aw2b  = (unsigned short*)(ws + AW2B_OFF);
    float*          gr    = (float*)(ws + GR_OFF);
    float*          out   = (float*)d_out;

    // re-zero accumulation buffers each call (deterministic, capture-legal)
    hipMemsetAsync(rst, 0, (size_t)N_NODES * HD * 4, stream);
    hipMemsetAsync(mk,  0, (size_t)N_NODES * HEADS * 4, stream);
    hipMemsetAsync(z,   0, (size_t)N_NODES * HEADS * 4, stream);

    k_prep<<<(65536 + 1024) / 256, 256, 0, stream>>>(fcw, aw1, aw2, fcwb, wcatb, aw2b);
    k_cvt_x<<<N_NODES * IN_FEATS / 4 / 256, 256, 0, stream>>>(x, xb);
    k_gemm_fc<<<12500, 128, 0, stream>>>(xb, fcwb, h);
    k_attn<<<(N_NODES * HEADS + 255) / 256, 256, 0, stream>>>(h, attl, attr, el, er);
    k_edge_e<<<N_EDGES / 256, 256, 0, stream>>>(src, dst, el, er, ews, mk);
    k_edge_exp<<<N_EDGES / 256, 256, 0, stream>>>(dst, ews, mk, z);
    k_msg<<<N_EDGES / 8, 256, 0, stream>>>(src, dst, ews, z, h, rst);
    k_elu<<<N_NODES * HD / 4 / 256, 256, 0, stream>>>(rst, gbias, h2b);
    k_colmean<<<HD, 256, 0, stream>>>(rst, gr);
    k_gemm_hsd<<<6250, 128, 0, stream>>>(h2b, wcatb, hsd);
    k_edge_actor<<<6250, 256, 0, stream>>>(src, dst, hsd, aw2b, ab1, ab2, out);
    k_critic<<<1, 64, 0, stream>>>(gr, cw1, cb1, cw2, cb2, out);
}
